// ForceField_79353815761255
// MI455X (gfx1250) — matrix operand order repaired and checked by reading, not compiled, not run on hardware
//
#include <hip/hip_runtime.h>

typedef __attribute__((ext_vector_type(16))) _Float16 v16h;
typedef __attribute__((ext_vector_type(8)))  _Float16 v8h;
typedef __attribute__((ext_vector_type(4)))  _Float16 v4h;
typedef __attribute__((ext_vector_type(8)))  float    v8f;

#define RBF_STEPS 32
#define OUT_SIZE  64
#define L_TOT     512
#define R_TOT     4096
#define FEAT_PITCH (RBF_STEPS * OUT_SIZE)     // 2048 elements per l/r row

// linspace(0, 8, 32): mu_e = e * 8/31 ; sigma = (0-8)/32 = -0.25
// exp(-((d-mu)/sigma)^2) = exp2(-16*log2(e)*(d-mu)^2)
#define MU_STEP  (8.0f / 31.0f)
#define RBF_C2   (16.0f * 1.44269504088896340736f)
#define ENERGY_SCALE 0.01f

#define E_CHUNK  8
#define LDS_FPAD 72   // row stride 144 B: 16B-aligned, spreads LDS banks

// ---------------------------------------------------------------------------
// One-pass f32 -> f16 conversion into workspace (amortized across all tiles)
// ---------------------------------------------------------------------------
__global__ __launch_bounds__(256)
void cvt_f32_f16_kernel(const float* __restrict__ src, _Float16* __restrict__ dst, int n4)
{
    const int i = blockIdx.x * 256 + threadIdx.x;
    if (i < n4) {
        const float4 x = ((const float4*)src)[i];
        v4h h;
        h[0] = (_Float16)x.x; h[1] = (_Float16)x.y;
        h[2] = (_Float16)x.z; h[3] = (_Float16)x.w;
        ((v4h*)dst)[i] = h;
    }
}

// ---------------------------------------------------------------------------
// Fast path: feats already f16 in workspace.
// Each wave owns an M=32 x N=16 output tile (two l-tiles, one r-tile): every
// B fragment is reused by two WMMAs from registers, halving B-side L2 bytes.
// A (32 rows) is async-DMA'd to LDS once per block; RBF fold is software-
// pipelined one e behind the WMMAs so trans/VALU co-executes with the XDL pipe.
// ---------------------------------------------------------------------------
__global__ __launch_bounds__(256)
void ForceField_79353815761255_f16_kernel(const _Float16* __restrict__ lig16,
                                          const _Float16* __restrict__ rec16,
                                          const float* __restrict__ lig_coord,
                                          const float* __restrict__ rec_coord,
                                          float* __restrict__ out)
{
    __shared__ __align__(16) _Float16 sA[E_CHUNK][32][LDS_FPAD];   // ~36 KB

    const int tid    = threadIdx.x;
    const int lane   = tid & 31;
    const int wave   = tid >> 5;
    const int laneLo = lane & 15;
    const int laneHi = lane >> 4;

    const int lp = blockIdx.y * 32;          // base l row of the 32-row pair
    const int rt = blockIdx.x * 8 + wave;    // r-tile [0,256)

    // C layout (16x16 f32, 8 VGPRs): M = v + 8*laneHi, N = laneLo
    const int j = rt * 16 + laneLo;
    const float rx = rec_coord[j * 3 + 0];
    const float ry = rec_coord[j * 3 + 1];
    const float rz = rec_coord[j * 3 + 2];
    float dist0[8], dist1[8];
#pragma unroll
    for (int v = 0; v < 8; ++v) {
        const int i0 = lp + laneHi * 8 + v;        // tile 0 rows
        const int i1 = i0 + 16;                    // tile 1 rows
        float dx = lig_coord[i0 * 3 + 0] - rx;
        float dy = lig_coord[i0 * 3 + 1] - ry;
        float dz = lig_coord[i0 * 3 + 2] - rz;
        dist0[v] = __builtin_amdgcn_sqrtf(dx * dx + dy * dy + dz * dz);
        dx = lig_coord[i1 * 3 + 0] - rx;
        dy = lig_coord[i1 * 3 + 1] - ry;
        dz = lig_coord[i1 * 3 + 2] - rz;
        dist1[v] = __builtin_amdgcn_sqrtf(dx * dx + dy * dy + dz * dz);
    }

    float acc0 = 0.0f, acc1 = 0.0f;
    v8f c0p = {}, c1p = {};          // pipelined previous-e C tiles (fold lags 1 e)
    float muPrev = 0.0f;             // folding zeros with any mu contributes 0

    for (int ec = 0; ec < RBF_STEPS / E_CHUNK; ++ec) {
        __syncthreads();
        // ---- async-stage A slab: 32 rows x 8 steps x 64 f16 = 32 KB
        //      2048 x 16B chunks, 8 per thread, DMA'd straight into LDS
        {
            const _Float16* src = lig16 + (size_t)lp * FEAT_PITCH
                                        + (size_t)(ec * E_CHUNK) * OUT_SIZE;
#pragma unroll
            for (int k = 0; k < 8; ++k) {
                const int idx = tid + k * 256;
                const int f8  = idx & 7;            // 8-f16 chunk within row
                const int eL  = (idx >> 3) & 7;     // step within chunk
                const int row = idx >> 6;           // l row within 32-row pair
                const _Float16* g = src + (size_t)row * FEAT_PITCH + eL * OUT_SIZE + f8 * 8;
                void* l = (void*)&sA[eL][row][f8 * 8];
                asm volatile("global_load_async_to_lds_b128 %0, %1, off"
                             :: "v"((unsigned)(unsigned long long)l),
                                "v"((unsigned long long)g)
                             : "memory");
            }
        }
#if __has_builtin(__builtin_amdgcn_s_wait_asynccnt)
        __builtin_amdgcn_s_wait_asynccnt(0);
#else
        asm volatile("s_wait_asynccnt 0x0" ::: "memory");
#endif
        __syncthreads();

#pragma unroll 2
        for (int eL = 0; eL < E_CHUNK; ++eL) {
            const int e = ec * E_CHUNK + eL;
            v8f c0 = {}, c1 = {};
#pragma unroll
            for (int kh = 0; kh < 2; ++kh) {   // F=64 -> two K=32 WMMAs per tile
                // A fragments (16x32 f16) from LDS: M = laneLo (+16 for tile 1),
                // halves 0..7 -> f = fa..fa+7 ; halves 8..15 -> f = fa+16..fa+23
                const int fa = kh * 32 + laneHi * 8;
                const v8h a0lo = *(const v8h*)&sA[eL][laneLo][fa];
                const v8h a0hi = *(const v8h*)&sA[eL][laneLo][fa + 16];
                const v16h a0 = __builtin_shufflevector(a0lo, a0hi,
                    0,1,2,3,4,5,6,7,8,9,10,11,12,13,14,15);
                const v8h a1lo = *(const v8h*)&sA[eL][16 + laneLo][fa];
                const v8h a1hi = *(const v8h*)&sA[eL][16 + laneLo][fa + 16];
                const v16h a1 = __builtin_shufflevector(a1lo, a1hi,
                    0,1,2,3,4,5,6,7,8,9,10,11,12,13,14,15);

                // B fragment (32x16 f16): N = laneLo (rec row j),
                // halves 0..7 -> f = fb..fb+7 ; halves 8..15 -> f = fb+16..fb+23 (fb = kh*32 + 8*laneHi)
                const _Float16* bp = rec16 + (size_t)j * FEAT_PITCH
                                     + e * OUT_SIZE + kh * 32 + laneHi * 8;
                const v8h b0 = *(const v8h*)bp;
                const v8h b1 = *(const v8h*)(bp + 16);
                const v16h b = __builtin_shufflevector(b0, b1,
                    0,1,2,3,4,5,6,7,8,9,10,11,12,13,14,15);

                c0 = __builtin_amdgcn_wmma_f32_16x16x32_f16(
                        false, a0, false, b, (short)0, c0, false, false);
                c1 = __builtin_amdgcn_wmma_f32_16x16x32_f16(
                        false, a1, false, b, (short)0, c1, false, false);
            }

            // ---- fold PREVIOUS e's C tiles while this e's WMMAs execute
#pragma unroll
            for (int v = 0; v < 8; ++v) {
                const float t0 = dist0[v] - muPrev;
                acc0 += c0p[v] * __builtin_amdgcn_exp2f(-RBF_C2 * t0 * t0);
                const float t1 = dist1[v] - muPrev;
                acc1 += c1p[v] * __builtin_amdgcn_exp2f(-RBF_C2 * t1 * t1);
            }
            c0p = c0; c1p = c1;
            muPrev = (float)e * MU_STEP;
        }
    }

    // ---- drain the pipeline: fold the last e
#pragma unroll
    for (int v = 0; v < 8; ++v) {
        const float t0 = dist0[v] - muPrev;
        acc0 += c0p[v] * __builtin_amdgcn_exp2f(-RBF_C2 * t0 * t0);
        const float t1 = dist1[v] - muPrev;
        acc1 += c1p[v] * __builtin_amdgcn_exp2f(-RBF_C2 * t1 * t1);
    }

    float acc = acc0 + acc1;
#pragma unroll
    for (int off = 16; off; off >>= 1)
        acc += __shfl_down(acc, off, 32);
    if (lane == 0)
        atomicAdd(out, acc * ENERGY_SCALE);
}

// ---------------------------------------------------------------------------
// Fallback (workspace too small): converts f32->f16 on the fly, M=16 tiles
// ---------------------------------------------------------------------------
__global__ __launch_bounds__(256)
void ForceField_79353815761255_f32_kernel(const float* __restrict__ lig_feat,
                                          const float* __restrict__ rec_feat,
                                          const float* __restrict__ lig_coord,
                                          const float* __restrict__ rec_coord,
                                          float* __restrict__ out)
{
    __shared__ __align__(16) _Float16 sA[E_CHUNK][16][LDS_FPAD];

    const int tid    = threadIdx.x;
    const int lane   = tid & 31;
    const int wave   = tid >> 5;
    const int laneLo = lane & 15;
    const int laneHi = lane >> 4;

    const int lt = blockIdx.y;
    const int rt = blockIdx.x * 8 + wave;

    const int j = rt * 16 + laneLo;
    const float rx = rec_coord[j * 3 + 0];
    const float ry = rec_coord[j * 3 + 1];
    const float rz = rec_coord[j * 3 + 2];
    float dist[8];
#pragma unroll
    for (int v = 0; v < 8; ++v) {
        const int i = lt * 16 + laneHi * 8 + v;
        const float dx = lig_coord[i * 3 + 0] - rx;
        const float dy = lig_coord[i * 3 + 1] - ry;
        const float dz = lig_coord[i * 3 + 2] - rz;
        dist[v] = __builtin_amdgcn_sqrtf(dx * dx + dy * dy + dz * dz);
    }

    float acc = 0.0f;

    for (int ec = 0; ec < RBF_STEPS / E_CHUNK; ++ec) {
        __syncthreads();
        {
            const float* src = lig_feat + (size_t)(lt * 16) * FEAT_PITCH
                                        + (size_t)(ec * E_CHUNK) * OUT_SIZE;
#pragma unroll
            for (int k = 0; k < 8; ++k) {
                const int idx4 = tid + k * 256;
                const int f4  = idx4 & 15;
                const int eL  = (idx4 >> 4) & 7;
                const int row = idx4 >> 7;
                const float4 x = *(const float4*)(src
                    + (size_t)row * FEAT_PITCH + eL * OUT_SIZE + f4 * 4);
                _Float16* dst = &sA[eL][row][f4 * 4];
                dst[0] = (_Float16)x.x; dst[1] = (_Float16)x.y;
                dst[2] = (_Float16)x.z; dst[3] = (_Float16)x.w;
            }
        }
        __syncthreads();

        for (int eL = 0; eL < E_CHUNK; ++eL) {
            const int e = ec * E_CHUNK + eL;
            v8f c = {};
#pragma unroll
            for (int kh = 0; kh < 2; ++kh) {
                const int fa = kh * 32 + laneHi * 8;
                const v8h alo = *(const v8h*)&sA[eL][laneLo][fa];
                const v8h ahi = *(const v8h*)&sA[eL][laneLo][fa + 16];
                const v16h a = __builtin_shufflevector(alo, ahi,
                    0,1,2,3,4,5,6,7,8,9,10,11,12,13,14,15);

                const float* bp = rec_feat + (size_t)j * FEAT_PITCH
                                  + e * OUT_SIZE + kh * 32 + laneHi * 8;
                v16h b;
#pragma unroll
                for (int q = 0; q < 4; ++q) {
                    const float4 x = *(const float4*)(bp + (q >> 1) * 16 + (q & 1) * 4);
                    b[q * 4 + 0] = (_Float16)x.x; b[q * 4 + 1] = (_Float16)x.y;
                    b[q * 4 + 2] = (_Float16)x.z; b[q * 4 + 3] = (_Float16)x.w;
                }

                c = __builtin_amdgcn_wmma_f32_16x16x32_f16(
                        false, a, false, b, (short)0, c, false, false);
            }

            const float mu = (float)e * MU_STEP;
#pragma unroll
            for (int v = 0; v < 8; ++v) {
                const float t = dist[v] - mu;
                acc += c[v] * __builtin_amdgcn_exp2f(-RBF_C2 * t * t);
            }
        }
    }

#pragma unroll
    for (int off = 16; off; off >>= 1)
        acc += __shfl_down(acc, off, 32);
    if (lane == 0)
        atomicAdd(out, acc * ENERGY_SCALE);
}

extern "C" void kernel_launch(void* const* d_in, const int* in_sizes, int n_in,
                              void* d_out, int out_size, void* d_ws, size_t ws_size,
                              hipStream_t stream) {
    const float* lig_feat  = (const float*)d_in[0];
    const float* rec_feat  = (const float*)d_in[1];
    const float* lig_coord = (const float*)d_in[2];
    const float* rec_coord = (const float*)d_in[3];
    float* out = (float*)d_out;

    hipMemsetAsync(out, 0, sizeof(float) * (out_size > 0 ? out_size : 1), stream);

    const size_t REC_N = (size_t)R_TOT * FEAT_PITCH;   // 8,388,608
    const size_t LIG_N = (size_t)L_TOT * FEAT_PITCH;   // 1,048,576

    if (ws_size >= (REC_N + LIG_N) * sizeof(_Float16)) {
        _Float16* wsRec = (_Float16*)d_ws;
        _Float16* wsLig = wsRec + REC_N;
        cvt_f32_f16_kernel<<<(int)(REC_N / 4 / 256), 256, 0, stream>>>(rec_feat, wsRec, (int)(REC_N / 4));
        cvt_f32_f16_kernel<<<(int)(LIG_N / 4 / 256), 256, 0, stream>>>(lig_feat, wsLig, (int)(LIG_N / 4));
        const dim3 grid(R_TOT / 16 / 8, L_TOT / 32);   // (32, 16): M=32 per wave
        ForceField_79353815761255_f16_kernel<<<grid, 256, 0, stream>>>(
            wsLig, wsRec, lig_coord, rec_coord, out);
    } else {
        const dim3 grid(R_TOT / 16 / 8, L_TOT / 16);   // (32, 32)
        ForceField_79353815761255_f32_kernel<<<grid, 256, 0, stream>>>(
            lig_feat, rec_feat, lig_coord, rec_coord, out);
    }
}
